// RNAGNN_9801115369587
// MI455X (gfx1250) — matrix operand order repaired and checked by reading, not compiled, not run on hardware
//
#include <hip/hip_runtime.h>

#define NN 20000
#define EE 400000
#define DD 256
#define HH 4
#define CC 64
#define LL 4

typedef __bf16 bf16_t;
typedef __attribute__((ext_vector_type(16))) __bf16 bfx16;
typedef __attribute__((ext_vector_type(8)))  __bf16 bfx8;
typedef __attribute__((ext_vector_type(8)))  float  fx8;

__device__ __forceinline__ void atomic_max_f32(float* addr, float val) {
    if (val >= 0.0f) atomicMax((int*)addr, __float_as_int(val));
    else             atomicMin((unsigned int*)addr, __float_as_uint(val));
}

// ---------------------------------------------------------------------------
// WMMA GEMM: C[M x 256] = A[M x 256](bf16, row-major) @ W[256 x 256] + bias
// W supplied pre-transposed as BT[n][k] (N x K, bf16). M must be multiple of 16.
// Block = 128 threads (4 waves). Block covers 16 rows x 256 cols.
// Each wave: 4 column tiles of 16; K-loop step 32 -> v_wmma_f32_16x16x32_bf16.
// ---------------------------------------------------------------------------
template <bool OUT_BF16>
__global__ void __launch_bounds__(128)
wmma_gemm_k256(const bf16_t* __restrict__ A, const bf16_t* __restrict__ BT,
               const float* __restrict__ bias, void* __restrict__ Cout, int M)
{
    const int lane   = threadIdx.x & 31;
    const int wave   = threadIdx.x >> 5;    // 0..3
    const int half   = lane >> 4;           // 0/1
    const int lsub   = lane & 15;
    const int rowBase = blockIdx.x << 4;
    if (rowBase >= M) return;               // wave-uniform

    const bf16_t* arow = A + (size_t)(rowBase + lsub) * DD;
    const int colBase = wave * 64;

    fx8 acc[4];
    #pragma unroll
    for (int t = 0; t < 4; ++t) acc[t] = (fx8){0.f,0.f,0.f,0.f,0.f,0.f,0.f,0.f};

    #pragma unroll
    for (int k0 = 0; k0 < DD; k0 += 32) {
        // A fragment (16x32 bf16): lane half 0 -> K {k0+0..7, k0+16..23},
        //                          lane half 1 -> K {k0+8..15, k0+24..31}
        union { bfx16 v; bfx8 p[2]; } af;
        af.p[0] = *(const bfx8*)(arow + k0 + half * 8);
        af.p[1] = *(const bfx8*)(arow + k0 + half * 8 + 16);
        #pragma unroll
        for (int t = 0; t < 4; ++t) {
            // B fragment (32x16 bf16): lane holds column (colBase+t*16+lsub),
            // K {k0+half*8+0..7, k0+half*8+16..23} (same order as A)
            const bf16_t* bcol = BT + (size_t)(colBase + t * 16 + lsub) * DD + k0 + half * 8;
            union { bfx16 v; bfx8 p[2]; } bfr;
            bfr.p[0] = *(const bfx8*)(bcol);
            bfr.p[1] = *(const bfx8*)(bcol + 16);
            acc[t] = __builtin_amdgcn_wmma_f32_16x16x32_bf16(
                false, af.v, false, bfr.v, (short)0, acc[t], false, false);
        }
    }

    // C/D layout: VGPR r, lanes 0-15 -> M=r, lanes 16-31 -> M=8+r; N = lsub
    #pragma unroll
    for (int t = 0; t < 4; ++t) {
        const int col = colBase + t * 16 + lsub;
        const float bb = bias ? bias[col] : 0.0f;
        const int mBase = rowBase + half * 8;
        #pragma unroll
        for (int r = 0; r < 8; ++r) {
            float val = acc[t][r] + bb;
            size_t o = (size_t)(mBase + r) * DD + col;
            if (OUT_BF16) ((bf16_t*)Cout)[o] = (bf16_t)val;
            else          ((float*)Cout)[o]  = val;
        }
    }
}

// ---------------------------------------------------------------------------
// Encoder: out = LN(relu(X @ W + b)) ; X is [M, IN_DIM], W [IN_DIM, 256]
// one block (256 threads) per row
// ---------------------------------------------------------------------------
template <int IN_DIM>
__global__ void __launch_bounds__(256)
encode_ln(const float* __restrict__ X, const float* __restrict__ W,
          const float* __restrict__ bias, const float* __restrict__ g,
          const float* __restrict__ be, float* __restrict__ outF,
          bf16_t* __restrict__ outB)
{
    __shared__ float xs[IN_DIM];
    __shared__ float red[DD];
    const int n = blockIdx.x;
    const int d = threadIdx.x;
    if (d < IN_DIM) xs[d] = X[(size_t)n * IN_DIM + d];
    __syncthreads();
    float s = bias[d];
    #pragma unroll
    for (int j = 0; j < IN_DIM; ++j) s = fmaf(xs[j], W[j * DD + d], s);
    s = fmaxf(s, 0.0f);

    red[d] = s; __syncthreads();
    for (int o = 128; o > 0; o >>= 1) { if (d < o) red[d] += red[d + o]; __syncthreads(); }
    const float m = red[0] * (1.0f / DD);
    __syncthreads();
    const float c = s - m;
    red[d] = c * c; __syncthreads();
    for (int o = 128; o > 0; o >>= 1) { if (d < o) red[d] += red[d + o]; __syncthreads(); }
    const float var = red[0] * (1.0f / DD);
    const float y = g[d] * c * rsqrtf(var + 1e-5f) + be[d];
    if (outF) outF[(size_t)n * DD + d] = y;
    if (outB) outB[(size_t)n * DD + d] = (bf16_t)y;
}

// ---------------------------------------------------------------------------
// y = LN(relu(in)); if resid: y += resid; if reluAfter: y = relu(y)
// ---------------------------------------------------------------------------
__global__ void __launch_bounds__(256)
ln_post(const float* __restrict__ in, const float* __restrict__ g,
        const float* __restrict__ be, const float* __restrict__ resid,
        int reluAfter, float* __restrict__ outF, bf16_t* __restrict__ outB)
{
    __shared__ float red[DD];
    const int n = blockIdx.x;
    const int d = threadIdx.x;
    float s = fmaxf(in[(size_t)n * DD + d], 0.0f);

    red[d] = s; __syncthreads();
    for (int o = 128; o > 0; o >>= 1) { if (d < o) red[d] += red[d + o]; __syncthreads(); }
    const float m = red[0] * (1.0f / DD);
    __syncthreads();
    const float c = s - m;
    red[d] = c * c; __syncthreads();
    for (int o = 128; o > 0; o >>= 1) { if (d < o) red[d] += red[d + o]; __syncthreads(); }
    const float var = red[0] * (1.0f / DD);
    float y = g[d] * c * rsqrtf(var + 1e-5f) + be[d];
    if (resid) y += resid[(size_t)n * DD + d];
    if (reluAfter) y = fmaxf(y, 0.0f);
    if (outF) outF[(size_t)n * DD + d] = y;
    if (outB) outB[(size_t)n * DD + d] = (bf16_t)y;
}

__global__ void seg_init(float* __restrict__ amax, float* __restrict__ denom)
{
    int idx = blockIdx.x * blockDim.x + threadIdx.x;
    if (idx >= NN * HH) return;
    amax[idx]  = -INFINITY;
    denom[idx] = 0.0f;
}

__global__ void attn_alpha(const float* __restrict__ q, const float* __restrict__ k,
                           const bf16_t* __restrict__ eeb, const int* __restrict__ src,
                           const int* __restrict__ dst, float* __restrict__ alpha,
                           float* __restrict__ amax)
{
    int idx = blockIdx.x * blockDim.x + threadIdx.x;
    if (idx >= EE * HH) return;
    const int e  = idx >> 2;
    const int hh = idx & 3;
    const int sN = src[e], dN = dst[e];
    const float*  qp = q   + (size_t)dN * DD + hh * CC;
    const float*  kp = k   + (size_t)sN * DD + hh * CC;
    const bf16_t* ep = eeb + (size_t)e  * DD + hh * CC;
    float s = 0.0f;
    #pragma unroll 8
    for (int c2 = 0; c2 < CC; ++c2) s = fmaf(qp[c2], kp[c2] + (float)ep[c2], s);
    s *= 0.125f;  // 1/sqrt(64)
    alpha[idx] = s;
    atomic_max_f32(&amax[dN * HH + hh], s);
}

__global__ void attn_exp(const int* __restrict__ dst, float* __restrict__ alpha,
                         const float* __restrict__ amax, float* __restrict__ denom)
{
    int idx = blockIdx.x * blockDim.x + threadIdx.x;
    if (idx >= EE * HH) return;
    const int e  = idx >> 2;
    const int hh = idx & 3;
    const int dN = dst[e];
    const float ex = expf(alpha[idx] - amax[dN * HH + hh]);
    alpha[idx] = ex;
    atomicAdd(&denom[dN * HH + hh], ex);
}

__global__ void attn_msg(const int* __restrict__ src, const int* __restrict__ dst,
                         const float* __restrict__ alpha, const float* __restrict__ denom,
                         const float* __restrict__ v, const bf16_t* __restrict__ eeb,
                         float* __restrict__ out)
{
    int idx = blockIdx.x * blockDim.x + threadIdx.x;  // EE*DD < 2^31
    if (idx >= EE * DD) return;
    const int e  = idx >> 8;
    const int d  = idx & 255;
    const int hh = d >> 6;
    const int sN = src[e], dN = dst[e];
    const float a = alpha[e * HH + hh] / denom[dN * HH + hh];
    const float m = a * (v[(size_t)sN * DD + d] + (float)eeb[(size_t)e * DD + d]);
    atomicAdd(&out[(size_t)dN * DD + d], m);
}

// W [256x256] f32 row-major -> WT[n*256+k] = bf16(W[k*256+n])
__global__ void wconv_t(const float* __restrict__ W, bf16_t* __restrict__ WT)
{
    int idx = blockIdx.x * 256 + threadIdx.x;   // 65536 total
    int nCol = idx >> 8, kk = idx & 255;
    WT[idx] = (bf16_t)W[kk * DD + nCol];
}

__global__ void f2bf(const float* __restrict__ s, bf16_t* __restrict__ d, int cnt)
{
    int idx = blockIdx.x * blockDim.x + threadIdx.x;
    if (idx < cnt) d[idx] = (bf16_t)s[idx];
}

__global__ void cls_head(const float* __restrict__ h1, const float* __restrict__ W2,
                         const float* __restrict__ b2, float* __restrict__ out)
{
    int idx = blockIdx.x * blockDim.x + threadIdx.x;  // NN*4
    if (idx >= NN * 4) return;
    const int n = idx >> 2, j = idx & 3;
    float s = b2[j];
    #pragma unroll 8
    for (int d2 = 0; d2 < DD; ++d2) s = fmaf(h1[(size_t)n * DD + d2], W2[d2 * 4 + j], s);
    out[idx] = s;
}

extern "C" void kernel_launch(void* const* d_in, const int* in_sizes, int n_in,
                              void* d_out, int out_size, void* d_ws, size_t ws_size,
                              hipStream_t stream)
{
    const float* x        = (const float*)d_in[0];
    const int*   ei       = (const int*)  d_in[1];
    const float* edge_at  = (const float*)d_in[2];
    const float* feat_W   = (const float*)d_in[3];
    const float* feat_b   = (const float*)d_in[4];
    const float* feat_g   = (const float*)d_in[5];
    const float* feat_be  = (const float*)d_in[6];
    const float* edge_W   = (const float*)d_in[7];
    const float* edge_b   = (const float*)d_in[8];
    const float* edge_g   = (const float*)d_in[9];
    const float* edge_be  = (const float*)d_in[10];
    const float* Wq       = (const float*)d_in[11];
    const float* bq       = (const float*)d_in[12];
    const float* Wk       = (const float*)d_in[13];
    const float* bk       = (const float*)d_in[14];
    const float* Wv       = (const float*)d_in[15];
    const float* bv       = (const float*)d_in[16];
    const float* We       = (const float*)d_in[17];
    const float* Wskip    = (const float*)d_in[18];
    const float* bskip    = (const float*)d_in[19];
    const float* skW      = (const float*)d_in[20];
    const float* skb      = (const float*)d_in[21];
    const float* skg      = (const float*)d_in[22];
    const float* skbe     = (const float*)d_in[23];
    const float* clsW1    = (const float*)d_in[24];
    const float* clsb1    = (const float*)d_in[25];
    const float* clsg     = (const float*)d_in[26];
    const float* clsbe    = (const float*)d_in[27];
    const float* clsW2    = (const float*)d_in[28];
    const float* clsb2    = (const float*)d_in[29];
    float* logits = (float*)d_out;
    const int* srcIdx = ei;
    const int* dstIdx = ei + EE;

    char* base = (char*)d_ws;
    size_t off = 0;
    auto alloc = [&](size_t bytes) -> void* {
        void* r = base + off;
        off = (off + bytes + 255) & ~(size_t)255;
        return r;
    };

    const size_t WSZ = (size_t)DD * DD;  // 65536
    bf16_t* WqT    = (bf16_t*)alloc(LL * WSZ * sizeof(bf16_t));
    bf16_t* WkT    = (bf16_t*)alloc(LL * WSZ * sizeof(bf16_t));
    bf16_t* WvT    = (bf16_t*)alloc(LL * WSZ * sizeof(bf16_t));
    bf16_t* WeT    = (bf16_t*)alloc(LL * WSZ * sizeof(bf16_t));
    bf16_t* WskipT = (bf16_t*)alloc(LL * WSZ * sizeof(bf16_t));
    bf16_t* skWT   = (bf16_t*)alloc(LL * WSZ * sizeof(bf16_t));
    bf16_t* clsW1T = (bf16_t*)alloc(WSZ * sizeof(bf16_t));

    float*  h     = (float*) alloc((size_t)NN * DD * sizeof(float));
    bf16_t* hbf   = (bf16_t*)alloc((size_t)NN * DD * sizeof(bf16_t));
    bf16_t* ebf   = (bf16_t*)alloc((size_t)EE * DD * sizeof(bf16_t));
    bf16_t* eebf  = (bf16_t*)alloc((size_t)EE * DD * sizeof(bf16_t));
    float*  qb    = (float*) alloc((size_t)NN * DD * sizeof(float));
    float*  kb    = (float*) alloc((size_t)NN * DD * sizeof(float));
    float*  vb    = (float*) alloc((size_t)NN * DD * sizeof(float));
    float*  outb  = (float*) alloc((size_t)NN * DD * sizeof(float));
    bf16_t* outbf = (bf16_t*)alloc((size_t)NN * DD * sizeof(bf16_t));
    float*  tmp   = (float*) alloc((size_t)NN * DD * sizeof(float));
    float*  alpha = (float*) alloc((size_t)EE * HH * sizeof(float));
    float*  amax  = (float*) alloc((size_t)NN * HH * sizeof(float));
    float*  denom = (float*) alloc((size_t)NN * HH * sizeof(float));

    // --- pre-convert weights to bf16, transposed to [N][K] for WMMA B-frags ---
    for (int i = 0; i < LL; ++i) {
        wconv_t<<<256, 256, 0, stream>>>(Wq    + i * WSZ, WqT    + i * WSZ);
        wconv_t<<<256, 256, 0, stream>>>(Wk    + i * WSZ, WkT    + i * WSZ);
        wconv_t<<<256, 256, 0, stream>>>(Wv    + i * WSZ, WvT    + i * WSZ);
        wconv_t<<<256, 256, 0, stream>>>(We    + i * WSZ, WeT    + i * WSZ);
        wconv_t<<<256, 256, 0, stream>>>(Wskip + i * WSZ, WskipT + i * WSZ);
        wconv_t<<<256, 256, 0, stream>>>(skW   + i * WSZ, skWT   + i * WSZ);
    }
    wconv_t<<<256, 256, 0, stream>>>(clsW1, clsW1T);

    // --- encoders ---
    encode_ln<34><<<NN, 256, 0, stream>>>(x, feat_W, feat_b, feat_g, feat_be, h, hbf);
    encode_ln<19><<<EE, 256, 0, stream>>>(edge_at, edge_W, edge_b, edge_g, edge_be,
                                          nullptr, ebf);

    // --- transformer layers ---
    for (int i = 0; i < LL; ++i) {
        const size_t wo = (size_t)i * WSZ;
        wmma_gemm_k256<false><<<NN / 16, 128, 0, stream>>>(hbf, WqT + wo, bq + i * DD, qb, NN);
        wmma_gemm_k256<false><<<NN / 16, 128, 0, stream>>>(hbf, WkT + wo, bk + i * DD, kb, NN);
        wmma_gemm_k256<false><<<NN / 16, 128, 0, stream>>>(hbf, WvT + wo, bv + i * DD, vb, NN);
        wmma_gemm_k256<true ><<<EE / 16, 128, 0, stream>>>(ebf, WeT + wo, nullptr, eebf, EE);

        seg_init<<<(NN * HH + 255) / 256, 256, 0, stream>>>(amax, denom);
        attn_alpha<<<(EE * HH + 255) / 256, 256, 0, stream>>>(qb, kb, eebf, srcIdx, dstIdx,
                                                              alpha, amax);
        attn_exp<<<(EE * HH + 255) / 256, 256, 0, stream>>>(dstIdx, alpha, amax, denom);

        // out = h @ Wskip + bskip, then scatter-add messages
        wmma_gemm_k256<false><<<NN / 16, 128, 0, stream>>>(hbf, WskipT + wo, bskip + i * DD,
                                                           outb, NN);
        attn_msg<<<(EE * DD) / 256, 256, 0, stream>>>(srcIdx, dstIdx, alpha, denom, vb,
                                                      eebf, outb);

        f2bf<<<(NN * DD + 255) / 256, 256, 0, stream>>>(outb, outbf, NN * DD);
        wmma_gemm_k256<false><<<NN / 16, 128, 0, stream>>>(outbf, skWT + wo, skb + i * DD,
                                                           tmp, NN);
        ln_post<<<NN, 256, 0, stream>>>(tmp, skg + i * DD, skbe + i * DD, h,
                                        (i < LL - 1) ? 1 : 0, h, hbf);
    }

    // --- classifier head ---
    wmma_gemm_k256<false><<<NN / 16, 128, 0, stream>>>(hbf, clsW1T, clsb1, tmp, NN);
    ln_post<<<NN, 256, 0, stream>>>(tmp, clsg, clsbe, nullptr, 0, outb, nullptr);
    cls_head<<<(NN * 4 + 255) / 256, 256, 0, stream>>>(outb, clsW2, clsb2, logits);
}
